// LinOSSLayer_53936199303325
// MI455X (gfx1250) — matrix operand order repaired and checked by reading, not compiled, not run on hardware
//
#include <hip/hip_runtime.h>

#define DEVINL __device__ __forceinline__

typedef __attribute__((ext_vector_type(16))) __bf16 bf16x16;
typedef __attribute__((ext_vector_type(8)))  float  f32x8;

// ---- fixed problem dims (from reference setup_inputs) ----
constexpr int BSZ  = 16;
constexpr int LSEQ = 2048;
constexpr int HDIM = 256;
constexpr int SDIM = 256;
constexpr int ROWS = BSZ * LSEQ;       // 32768 (b*L rows)
constexpr int K1   = HDIM;             // 256
constexpr int N1   = 2 * SDIM;         // 512   bu columns: [real(s) | imag(s)]
constexpr int K2   = 2 * SDIM;         // 512
constexpr int N2   = HDIM;             // 256
constexpr int CHUNK = 128;
constexpr int NCH   = LSEQ / CHUNK;    // 16

// ---- workspace layout (bytes) ----
constexpr size_t OFF_INBF = 0;                                   // ROWS*K1 bf16
constexpr size_t SZ_INBF  = (size_t)ROWS * K1 * 2;
constexpr size_t OFF_W1T  = OFF_INBF + SZ_INBF;                  // N1 x K1 bf16
constexpr size_t SZ_W1T   = (size_t)N1 * K1 * 2;
constexpr size_t OFF_W2T  = OFF_W1T + SZ_W1T;                    // N2 x K2 bf16
constexpr size_t SZ_W2T   = (size_t)N2 * K2 * 2;
constexpr size_t OFF_SCAL = OFF_W2T + SZ_W2T;                    // SDIM * 16 f32
constexpr size_t SZ_SCAL  = (size_t)SDIM * 16 * 4;
constexpr size_t OFF_BU   = OFF_SCAL + SZ_SCAL;                  // ROWS x N1 f32
constexpr size_t SZ_BU    = (size_t)ROWS * N1 * 4;
constexpr size_t OFF_X    = OFF_BU + SZ_BU;                      // ROWS x K2 bf16
constexpr size_t SZ_X     = (size_t)ROWS * K2 * 2;
constexpr size_t OFF_ZEND = OFF_X + SZ_X;                        // BSZ*2*NCH*SDIM float2
constexpr size_t SZ_ZEND  = (size_t)BSZ * 2 * NCH * SDIM * 2 * 4;
constexpr size_t OFF_CARR = OFF_ZEND + SZ_ZEND;                  // same size
// total ~114.5 MiB

// =====================================================================
// Fragment loaders (CDNA5 WMMA bf16 16x16x32 layouts, ISA 7.12.2)
// =====================================================================
union FragU { bf16x16 v; uint4 q[2]; };

// A operand 16x32 (MxK): lane<16 -> M=lane, K runs {k0+0..7},{k0+16..23};
// lane>=16 -> M=lane-16, K runs {k0+8..15},{k0+24..31}. Two 16B contiguous runs.
DEVINL bf16x16 load_frag_a(const __bf16* __restrict__ A, int ld, int row0, int k0) {
  int lane = threadIdx.x & 31;
  int r = lane & 15, h = lane >> 4;
  const __bf16* p = A + (size_t)(row0 + r) * ld + k0 + h * 8;
  FragU f;
  f.q[0] = *reinterpret_cast<const uint4*>(p);
  f.q[1] = *reinterpret_cast<const uint4*>(p + 16);
  return f.v;
}

// B operand 32x16 (KxN) sourced column-major from Wt[N][K]:
// lane<16 -> N=lane, K runs {k0+0..7},{k0+16..23}; lane>=16 -> N=lane-16, K runs {k0+8..15},{k0+24..31}.
DEVINL bf16x16 load_frag_b(const __bf16* __restrict__ Wt, int ld, int n0, int k0) {
  int lane = threadIdx.x & 31;
  int n = lane & 15, h = lane >> 4;
  const __bf16* p = Wt + (size_t)(n0 + n) * ld + k0 + h * 8;
  FragU f;
  f.q[0] = *reinterpret_cast<const uint4*>(p);
  f.q[1] = *reinterpret_cast<const uint4*>(p + 16);
  return f.v;
}

DEVINL f32x8 wmma_bf16(bf16x16 a, bf16x16 b, f32x8 c) {
  return __builtin_amdgcn_wmma_f32_16x16x32_bf16(false, a, false, b, (short)0, c,
                                                 false, false);
}

// =====================================================================
// GEMM: out[M x NCOLS] = A[M x KTOT](bf16) * Wt^T (Wt: NCOLS x KTOT bf16),
// fp32 accumulate. Wave computes 64x16 strip (4 WMMA tiles, B frag reused 4x).
// Software-pipelined k-loop; sched_barrier(0) pins the double-buffer distance
// (prevents the scheduler from sinking next-step loads below current WMMAs),
// so waits become partial (s_wait_loadcnt ~10) instead of full drains.
// EPI: out += addIn * Dvec (final projection epilogue).
// =====================================================================
template <int KTOT, int NCOLS, bool EPI>
__global__ void __launch_bounds__(256)
gemm_wmma(const __bf16* __restrict__ A, const __bf16* __restrict__ Wt,
          float* __restrict__ out,
          const float* __restrict__ addIn, const float* __restrict__ Dvec) {
  constexpr int NTN = NCOLS / 16;
  int wave  = (int)((blockIdx.x * blockDim.x + threadIdx.x) >> 5);
  int mt    = wave / NTN;          // 64-row strip index
  int nt    = wave % NTN;
  int mbase = mt * 64;
  int nbase = nt * 16;
  if (mbase >= ROWS) return;

  f32x8 acc[4] = {};

  // prologue: k-step 0 fragments
  bf16x16 bc  = load_frag_b(Wt, KTOT, nbase, 0);
  bf16x16 ac0 = load_frag_a(A, KTOT, mbase +  0, 0);
  bf16x16 ac1 = load_frag_a(A, KTOT, mbase + 16, 0);
  bf16x16 ac2 = load_frag_a(A, KTOT, mbase + 32, 0);
  bf16x16 ac3 = load_frag_a(A, KTOT, mbase + 48, 0);

#pragma unroll
  for (int k0 = 32; k0 < KTOT; k0 += 32) {
    // prefetch next k-step (10 b128 loads) before consuming current one
    bf16x16 bn  = load_frag_b(Wt, KTOT, nbase, k0);
    bf16x16 an0 = load_frag_a(A, KTOT, mbase +  0, k0);
    bf16x16 an1 = load_frag_a(A, KTOT, mbase + 16, k0);
    bf16x16 an2 = load_frag_a(A, KTOT, mbase + 32, k0);
    bf16x16 an3 = load_frag_a(A, KTOT, mbase + 48, k0);

    // keep the prefetch above the WMMAs: nothing may cross this point
    __builtin_amdgcn_sched_barrier(0);

    acc[0] = wmma_bf16(ac0, bc, acc[0]);
    acc[1] = wmma_bf16(ac1, bc, acc[1]);
    acc[2] = wmma_bf16(ac2, bc, acc[2]);
    acc[3] = wmma_bf16(ac3, bc, acc[3]);

    __builtin_amdgcn_sched_barrier(0);

    bc = bn; ac0 = an0; ac1 = an1; ac2 = an2; ac3 = an3;
  }
  // epilogue k-step
  acc[0] = wmma_bf16(ac0, bc, acc[0]);
  acc[1] = wmma_bf16(ac1, bc, acc[1]);
  acc[2] = wmma_bf16(ac2, bc, acc[2]);
  acc[3] = wmma_bf16(ac3, bc, acc[3]);

  int lane = threadIdx.x & 31;
  int r = lane & 15, h = lane >> 4;
#pragma unroll
  for (int i = 0; i < 4; ++i) {
#pragma unroll
    for (int g = 0; g < 8; ++g) {
      int row = mbase + i * 16 + h * 8 + g;
      int col = nbase + r;
      float v = acc[i][g];
      if (EPI) v += addIn[(size_t)row * NCOLS + col] * Dvec[col];
      out[(size_t)row * NCOLS + col] = v;
    }
  }
}

// =====================================================================
// Packing / scalar-prep kernels
// =====================================================================
union BF4 { __bf16 b[4]; uint2 u; };

__global__ void pack_inputs_bf16(const float* __restrict__ in, __bf16* __restrict__ out,
                                 int n4) {
  for (int i = blockIdx.x * blockDim.x + threadIdx.x; i < n4; i += gridDim.x * blockDim.x) {
    float4 f = reinterpret_cast<const float4*>(in)[i];
    BF4 o;
    o.b[0] = (__bf16)f.x; o.b[1] = (__bf16)f.y;
    o.b[2] = (__bf16)f.z; o.b[3] = (__bf16)f.w;
    reinterpret_cast<uint2*>(out)[i] = o.u;
  }
}

// W1t[(c*S + s)][h] = B[s][h][c]   (B is (S,H,2) row-major)
__global__ void pack_w1(const float* __restrict__ B, __bf16* __restrict__ W1t) {
  int idx = blockIdx.x * blockDim.x + threadIdx.x;     // over N1*K1
  if (idx >= N1 * K1) return;
  int nrow = idx / K1, hcol = idx % K1;
  int c = nrow >> 8, s = nrow & 255;
  W1t[idx] = (__bf16)B[((size_t)s * HDIM + hcol) * 2 + c];
}

// W2t[h][c*S + s] = c==0 ?  C[h][s][0] : -C[h][s][1]   (C is (H,S,2) row-major)
__global__ void pack_w2(const float* __restrict__ C, __bf16* __restrict__ W2t) {
  int idx = blockIdx.x * blockDim.x + threadIdx.x;     // over N2*K2
  if (idx >= N2 * K2) return;
  int hrow = idx / K2, k = idx % K2;
  int c = k >> 8, s = k & 255;
  float v = C[((size_t)hrow * SDIM + s) * 2 + c];
  W2t[idx] = (__bf16)(c ? -v : v);
}

// per-s scalars: m11,m12,m21,m22, f1c=m11*step, f2c=m21*step, P = M^CHUNK
__global__ void prep_scalars(const float* __restrict__ A_diag,
                             const float* __restrict__ steps,
                             float* __restrict__ scal) {
  int s = threadIdx.x;
  float a     = fmaxf(A_diag[s], 0.0f);
  float st    = 1.0f / (1.0f + expf(-steps[s]));
  float schur = 1.0f / (1.0f + st * st * a);
  float m11 = 1.0f - st * st * a * schur;
  float m12 = -st * a * schur;
  float m21 = st * schur;
  float m22 = schur;
  float p11 = 1.f, p12 = 0.f, p21 = 0.f, p22 = 1.f;
  for (int i = 0; i < CHUNK; ++i) {   // P <- M * P
    float q11 = m11 * p11 + m12 * p21;
    float q12 = m11 * p12 + m12 * p22;
    float q21 = m21 * p11 + m22 * p21;
    float q22 = m21 * p12 + m22 * p22;
    p11 = q11; p12 = q12; p21 = q21; p22 = q22;
  }
  float* o = scal + (size_t)s * 16;
  o[0] = m11; o[1] = m12; o[2] = m21; o[3] = m22;
  o[4] = m11 * st; o[5] = m21 * st;
  o[6] = p11; o[7] = p12; o[8] = p21; o[9] = p22;
}

// =====================================================================
// Scan: z_l = M z_{l-1} + (f1c,f2c)*bu_l  per (b,c,s); x_l = z_l[1].
// Pass 1: chunk-local end states.  Pass 2: carry sweep with P=M^CHUNK.
// Pass 3: re-run chunks from carries, emit x in bf16 at [row][c*S+s].
// =====================================================================
__global__ void __launch_bounds__(256)
scan_partial(const float* __restrict__ bu, const float* __restrict__ scal,
             float* __restrict__ zend) {
  int s  = threadIdx.x;
  int ch = blockIdx.x % NCH;
  int bc = blockIdx.x / NCH;                   // b*2 + c
  const float* sc = scal + (size_t)s * 16;
  float m11 = sc[0], m12 = sc[1], m21 = sc[2], m22 = sc[3], f1c = sc[4], f2c = sc[5];
  const float* base = bu + ((size_t)((bc >> 1) * LSEQ + ch * CHUNK)) * N1 + (bc & 1) * SDIM + s;
  float z1 = 0.f, z2 = 0.f;
  for (int j = 0; j < CHUNK; ++j) {
    float u = base[(size_t)j * N1];
    float n1 = m11 * z1 + m12 * z2 + u * f1c;
    float n2 = m21 * z1 + m22 * z2 + u * f2c;
    z1 = n1; z2 = n2;
  }
  size_t idx = ((size_t)bc * NCH + ch) * SDIM + s;
  zend[idx * 2]     = z1;
  zend[idx * 2 + 1] = z2;
}

__global__ void __launch_bounds__(256)
scan_carry(const float* __restrict__ scal, const float* __restrict__ zend,
           float* __restrict__ carry) {
  int s  = threadIdx.x;
  int bc = blockIdx.x;
  const float* sc = scal + (size_t)s * 16;
  float p11 = sc[6], p12 = sc[7], p21 = sc[8], p22 = sc[9];
  float c1 = 0.f, c2 = 0.f;
  for (int ch = 0; ch < NCH; ++ch) {
    size_t idx = ((size_t)bc * NCH + ch) * SDIM + s;
    carry[idx * 2]     = c1;
    carry[idx * 2 + 1] = c2;
    float e1 = zend[idx * 2], e2 = zend[idx * 2 + 1];
    float n1 = p11 * c1 + p12 * c2 + e1;
    float n2 = p21 * c1 + p22 * c2 + e2;
    c1 = n1; c2 = n2;
  }
}

__global__ void __launch_bounds__(256)
scan_final(const float* __restrict__ bu, const float* __restrict__ scal,
           const float* __restrict__ carry, __bf16* __restrict__ x) {
  int s  = threadIdx.x;
  int ch = blockIdx.x % NCH;
  int bc = blockIdx.x / NCH;
  const float* sc = scal + (size_t)s * 16;
  float m11 = sc[0], m12 = sc[1], m21 = sc[2], m22 = sc[3], f1c = sc[4], f2c = sc[5];
  size_t cidx = (((size_t)bc * NCH + ch) * SDIM + s) * 2;
  float z1 = carry[cidx], z2 = carry[cidx + 1];
  size_t row0 = (size_t)(bc >> 1) * LSEQ + ch * CHUNK;
  size_t colo = (size_t)(bc & 1) * SDIM + s;
  const float* base = bu + row0 * N1 + colo;
  __bf16* xo = x + row0 * K2 + colo;
  for (int j = 0; j < CHUNK; ++j) {
    float u = base[(size_t)j * N1];
    float n1 = m11 * z1 + m12 * z2 + u * f1c;
    float n2 = m21 * z1 + m22 * z2 + u * f2c;
    z1 = n1; z2 = n2;
    xo[(size_t)j * K2] = (__bf16)n2;
  }
}

// =====================================================================
// Launch
// =====================================================================
extern "C" void kernel_launch(void* const* d_in, const int* in_sizes, int n_in,
                              void* d_out, int out_size, void* d_ws, size_t ws_size,
                              hipStream_t stream) {
  const float* inputs = (const float*)d_in[0];
  const float* A_diag = (const float*)d_in[1];
  const float* Bw     = (const float*)d_in[2];
  const float* Cw     = (const float*)d_in[3];
  const float* Dw     = (const float*)d_in[4];
  const float* steps  = (const float*)d_in[5];
  float* out = (float*)d_out;

  char* ws = (char*)d_ws;
  __bf16* inBF  = (__bf16*)(ws + OFF_INBF);
  __bf16* W1t   = (__bf16*)(ws + OFF_W1T);
  __bf16* W2t   = (__bf16*)(ws + OFF_W2T);
  float*  scal  = (float*)(ws + OFF_SCAL);
  float*  bu    = (float*)(ws + OFF_BU);
  __bf16* xbuf  = (__bf16*)(ws + OFF_X);
  float*  zend  = (float*)(ws + OFF_ZEND);
  float*  carr  = (float*)(ws + OFF_CARR);

  // 1) pack operands to bf16
  pack_inputs_bf16<<<2048, 256, 0, stream>>>(inputs, inBF, (ROWS * K1) / 4);
  pack_w1<<<(N1 * K1) / 256, 256, 0, stream>>>(Bw, W1t);
  pack_w2<<<(N2 * K2) / 256, 256, 0, stream>>>(Cw, W2t);
  prep_scalars<<<1, 256, 0, stream>>>(A_diag, steps, scal);

  // 2) GEMM1: bu = inputs @ [B_r^T | B_i^T]   (32768x256)x(256x512)
  {
    int waves = (ROWS / 64) * (N1 / 16);
    gemm_wmma<K1, N1, false><<<waves / 8, 256, 0, stream>>>(inBF, W1t, bu, nullptr, nullptr);
  }

  // 3) chunked linear recurrence over L
  scan_partial<<<BSZ * 2 * NCH, 256, 0, stream>>>(bu, scal, zend);
  scan_carry<<<BSZ * 2, 256, 0, stream>>>(scal, zend, carr);
  scan_final<<<BSZ * 2 * NCH, 256, 0, stream>>>(bu, scal, carr, xbuf);

  // 4) GEMM2: out = x @ [C_r ; -C_i]^T + inputs*D   (32768x512)x(512x256)
  {
    int waves = (ROWS / 64) * (N2 / 16);
    gemm_wmma<K2, N2, true><<<waves / 8, 256, 0, stream>>>(xbuf, W2t, out, inputs, Dw);
  }
  (void)in_sizes; (void)n_in; (void)out_size; (void)ws_size;
}
